// PathEmbedding_26577257628306
// MI455X (gfx1250) — matrix operand order repaired and checked by reading, not compiled, not run on hardware
//
#include <hip/hip_runtime.h>
#include <hip/hip_bf16.h>

namespace {

constexpr int NUM_QUESTS = 4096;
constexpr int NUM_PATHS  = 4;
constexpr int LINK_DIM   = 64;
constexpr int PATH_LEN   = 16;
constexpr int B_PATHS    = NUM_QUESTS * NUM_PATHS;   // 16384
constexpr int WAVES      = 4;
constexpr int TPB        = WAVES * 32;               // 128 threads, wave32

typedef __attribute__((ext_vector_type(16))) _Float16 v16h;
typedef __attribute__((ext_vector_type(8)))  _Float16 v8h;
typedef __attribute__((ext_vector_type(8)))  float    v8f;
typedef __attribute__((ext_vector_type(4)))  float    v4f;

__device__ __forceinline__ v8f wmma_f16(v16h a, v16h b, v8f c) {
  // D = A(16x32 f16) * B(32x16 f16) + C(16x16 f32)
  return __builtin_amdgcn_wmma_f32_16x16x32_f16(false, a, false, b, (short)0, c,
                                                false, false);
}

__device__ __forceinline__ v16h cat16(v8h lo, v8h hp) {
  return __builtin_shufflevector(lo, hp, 0, 1, 2, 3, 4, 5, 6, 7, 8, 9, 10, 11,
                                 12, 13, 14, 15);
}

// A fragment (16x32 f16) from LDS row-major [16][64].
// Lane layout: m = lane&15, hi = lane>>4; halfs 0..7 -> k = kbase+hi*8+e,
// halfs 8..15 -> k = kbase+16+hi*8+(e-8). Two contiguous 16B runs => ds_load_b128.
__device__ __forceinline__ v16h ldA_lds(const _Float16* base, int kbase, int m, int hi) {
  const _Float16* p = base + m * 64 + kbase + hi * 8;
  v8h lo = *(const v8h*)(p);
  v8h hp = *(const v8h*)(p + 16);
  return cat16(lo, hp);
}

// A fragment gathered from a global f32 row of 64 (converted to f16).
__device__ __forceinline__ v16h ldA_g32(const float* row, int kbase, int hi) {
  const float* p = row + kbase + hi * 8;
  v4f a0 = *(const v4f*)(p);
  v4f a1 = *(const v4f*)(p + 4);
  v4f a2 = *(const v4f*)(p + 16);
  v4f a3 = *(const v4f*)(p + 20);
  v16h r;
#pragma unroll
  for (int i = 0; i < 4; ++i) {
    r[i]      = (_Float16)a0[i];
    r[4 + i]  = (_Float16)a1[i];
    r[8 + i]  = (_Float16)a2[i];
    r[12 + i] = (_Float16)a3[i];
  }
  return r;
}

// B fragment (32x16 f16) from LDS weight stored transposed: wt[n][k], k-stride 64.
// Lane layout: n = lane&15, halfs 0..7 -> k = ktile*32 + hi*8 + e, halfs 8..15 -> k = ktile*32 + 16 + hi*8 + (e-8).
__device__ __forceinline__ v16h ldB_lds(const _Float16* wt, int ntile, int ktile,
                                        int lane15, int hi) {
  const _Float16* p = wt + (ntile * 16 + lane15) * 64 + ktile * 32 + hi * 8;
  v8h lo = *(const v8h*)(p);
  v8h hp = *(const v8h*)(p + 16);
  return cat16(lo, hp);
}

__device__ __forceinline__ float sigm(float x) { return 1.0f / (1.0f + __expf(-x)); }

__device__ __forceinline__ float ftanh(float x) {
  x = fminf(fmaxf(x, -15.0f), 15.0f);
  float e = __expf(-2.0f * x);
  return (1.0f - e) / (1.0f + e);
}

__device__ __forceinline__ v8f splat8(float v) {
  v8f c;
#pragma unroll
  for (int i = 0; i < 8; ++i) c[i] = v;
  return c;
}

// One GRU sweep over the 16 steps for this wave's 16 paths.
// C-layout elementwise state: hprev[f][r] holds (row m=r+hi*8, dim n=f*16+lane15).
// Every step writes h to stag ([16 rows][64 dims] f16) so the next step can read
// it back in A-matrix layout (wave-local; LDS ops are in-order per wave).
// If ATT: accumulate sacc += (h_t . u) * h_t using 16-lane shuffle reductions.
template <bool ATT>
__device__ __forceinline__ void gru_pass(
    const float* __restrict__ link_states, const int* __restrict__ paths,
    const _Float16* __restrict__ sWxT, const _Float16* __restrict__ sWhT,
    _Float16* __restrict__ stag, const float* __restrict__ bz,
    int pbase, int m, int hi, int lane15,
    v8f hprev[4], const v8f accU[4], v8f sacc[4]) {
#pragma unroll 1
  for (int t = 0; t < PATH_LEN; ++t) {
    const bool first = (t == 0);
    const int edge = paths[(pbase + m) * PATH_LEN + t];
    const float* xrow = link_states + edge * LINK_DIM;
    v16h ax0 = ldA_g32(xrow, 0, hi);
    v16h ax1 = ldA_g32(xrow, 32, hi);
    v16h ah0, ah1;
    if (!first) {
      ah0 = ldA_lds(stag, 0, m, hi);
      ah1 = ldA_lds(stag, 32, m, hi);
    }
#pragma unroll
    for (int f = 0; f < 4; ++f) {
      // z gate (n-tiles 0..3), candidate gx part (n-tiles 8..11)
      v8f cz = splat8(bz[f]);
      cz = wmma_f16(ax0, ldB_lds(sWxT, f, 0, lane15, hi), cz);
      cz = wmma_f16(ax1, ldB_lds(sWxT, f, 1, lane15, hi), cz);
      v8f cx = splat8(bz[8 + f]);
      cx = wmma_f16(ax0, ldB_lds(sWxT, 8 + f, 0, lane15, hi), cx);
      cx = wmma_f16(ax1, ldB_lds(sWxT, 8 + f, 1, lane15, hi), cx);
      v8f cr = splat8(bz[4 + f]);
      v8f chh = splat8(0.0f);
      if (!first) {  // wave-uniform branch: EXEC stays all-ones around WMMA
        cz  = wmma_f16(ah0, ldB_lds(sWhT, f, 0, lane15, hi), cz);
        cz  = wmma_f16(ah1, ldB_lds(sWhT, f, 1, lane15, hi), cz);
        cr  = wmma_f16(ax0, ldB_lds(sWxT, 4 + f, 0, lane15, hi), cr);
        cr  = wmma_f16(ax1, ldB_lds(sWxT, 4 + f, 1, lane15, hi), cr);
        cr  = wmma_f16(ah0, ldB_lds(sWhT, 4 + f, 0, lane15, hi), cr);
        cr  = wmma_f16(ah1, ldB_lds(sWhT, 4 + f, 1, lane15, hi), cr);
        chh = wmma_f16(ah0, ldB_lds(sWhT, 8 + f, 0, lane15, hi), chh);
        chh = wmma_f16(ah1, ldB_lds(sWhT, 8 + f, 1, lane15, hi), chh);
      }
      v8f hn;
#pragma unroll
      for (int r = 0; r < 8; ++r) {
        float z  = sigm(cz[r]);
        float hh = ftanh(cx[r] + sigm(cr[r]) * chh[r]);  // chh==0 at t==0
        float hv = z * hprev[f][r] + (1.0f - z) * hh;    // hprev==0 at t==0
        hn[r] = hv;
        stag[(r + hi * 8) * 64 + f * 16 + lane15] = (_Float16)hv;
      }
      hprev[f] = hn;  // hprev now holds h_t in C layout (also used by ATT below)
    }
    if (ATT) {
#pragma unroll
      for (int r = 0; r < 8; ++r) {
        float p = hprev[0][r] * accU[0][r] + hprev[1][r] * accU[1][r] +
                  hprev[2][r] * accU[2][r] + hprev[3][r] * accU[3][r];
        p += __shfl_xor(p, 1);
        p += __shfl_xor(p, 2);
        p += __shfl_xor(p, 4);
        p += __shfl_xor(p, 8);  // att[row r, t] broadcast to all 16 lanes of group
#pragma unroll
        for (int f = 0; f < 4; ++f) sacc[f][r] += p * hprev[f][r];
      }
    }
  }
}

__global__ __launch_bounds__(TPB) void path_embed_kernel(
    const float* __restrict__ link_states, const float* __restrict__ Wx,
    const float* __restrict__ Wh, const float* __restrict__ bias,
    const float* __restrict__ wq, const float* __restrict__ wk,
    const float* __restrict__ wv, const int* __restrict__ paths,
    float* __restrict__ out) {
  __shared__ _Float16 sWxT[192 * 64];             // Wx^T : [n][k], f16 (24 KB)
  __shared__ _Float16 sWhT[192 * 64];             // Wh^T : [n][k], f16 (24 KB)
  __shared__ _Float16 sAux[64 * 64];              // Mkq, later wv^T   (8 KB)
  __shared__ _Float16 sStagAll[WAVES][16 * 64];   // per-wave h staging (8 KB)

  const int tid    = threadIdx.x;
  const int lane   = tid & 31;
  const int wave   = tid >> 5;
  const int lane15 = lane & 15;
  const int hi     = lane >> 4;
  const int m      = lane15;                       // A-layout row for this lane
  const int pbase  = (blockIdx.x * WAVES + wave) * 16;
  _Float16* stag   = sStagAll[wave];

  // --- cooperative weight prep (global reads coalesced, LDS writes scattered) ---
  // sWxT[n*64+k] = Wx[k*192+n]; iterate i over Wx's own layout so reads stream.
  for (int i = tid; i < 192 * 64; i += TPB) {
    int k = i / 192, n = i % 192;
    sWxT[n * 64 + k] = (_Float16)Wx[i];
    sWhT[n * 64 + k] = (_Float16)Wh[i];
  }
  // Mkq[d][e] = sum_n wk[d][n] * wq[e][n]  (row-major Mkq == B-layout for u-GEMM)
  for (int i = tid; i < 64 * 64; i += TPB) {
    int d = i >> 6, e = i & 63;
    const float* kr = wk + d * 64;
    const float* qr = wq + e * 64;
    float acc = 0.0f;
#pragma unroll 8
    for (int n = 0; n < 64; ++n) acc += kr[n] * qr[n];
    sAux[i] = (_Float16)acc;
  }
  __syncthreads();

  float bz[12];
#pragma unroll
  for (int j = 0; j < 12; ++j) bz[j] = bias[j * 16 + lane15];

  const v8f zero8 = splat8(0.0f);
  v8f hprev[4] = {zero8, zero8, zero8, zero8};
  v8f sacc[4]  = {zero8, zero8, zero8, zero8};
  v8f accU[4]  = {zero8, zero8, zero8, zero8};

  // pass 1: run GRU; final state h_L is left in stag
  gru_pass<false>(link_states, paths, sWxT, sWhT, stag, bz, pbase, m, hi, lane15,
                  hprev, accU, sacc);

  // u = h_L @ Mkq^T  (B read from row-major Mkq)
  {
    v16h a0 = ldA_lds(stag, 0, m, hi);
    v16h a1 = ldA_lds(stag, 32, m, hi);
#pragma unroll
    for (int f = 0; f < 4; ++f) {
      v8f c   = wmma_f16(a0, ldB_lds(sAux, f, 0, lane15, hi), zero8);
      accU[f] = wmma_f16(a1, ldB_lds(sAux, f, 1, lane15, hi), c);
    }
  }

  __syncthreads();
  // retarget aux region: sAux[n*64+k] = wv[k*64+n]; stream wv, scatter to LDS.
  for (int i = tid; i < 64 * 64; i += TPB) {
    int k = i >> 6, n = i & 63;
    sAux[n * 64 + k] = (_Float16)wv[i];
  }
  __syncthreads();

  // pass 2: recompute GRU, accumulate s = sum_t (h_t . u) h_t
#pragma unroll
  for (int f = 0; f < 4; ++f) hprev[f] = zero8;
  gru_pass<true>(link_states, paths, sWxT, sWhT, stag, bz, pbase, m, hi, lane15,
                 hprev, accU, sacc);

  // context = s @ wv  (transpose s through staging, then 8 WMMAs)
#pragma unroll
  for (int f = 0; f < 4; ++f) {
#pragma unroll
    for (int r = 0; r < 8; ++r)
      stag[(r + hi * 8) * 64 + f * 16 + lane15] = (_Float16)sacc[f][r];
  }
  v8f ctx[4];
  {
    v16h s0 = ldA_lds(stag, 0, m, hi);
    v16h s1 = ldA_lds(stag, 32, m, hi);
#pragma unroll
    for (int f = 0; f < 4; ++f) {
      v8f c  = wmma_f16(s0, ldB_lds(sAux, f, 0, lane15, hi), zero8);
      ctx[f] = wmma_f16(s1, ldB_lds(sAux, f, 1, lane15, hi), c);
    }
  }

  // softmax per quest: rows {4g..4g+3} of this half x 64 dims (16 lanes x 4 frags)
#pragma unroll
  for (int g = 0; g < 2; ++g) {
    float mx = -3.4e38f;
#pragma unroll
    for (int f = 0; f < 4; ++f)
#pragma unroll
      for (int rr = 0; rr < 4; ++rr) mx = fmaxf(mx, ctx[f][g * 4 + rr]);
    mx = fmaxf(mx, __shfl_xor(mx, 1));
    mx = fmaxf(mx, __shfl_xor(mx, 2));
    mx = fmaxf(mx, __shfl_xor(mx, 4));
    mx = fmaxf(mx, __shfl_xor(mx, 8));
    float ex[4][4];
    float se = 0.0f;
#pragma unroll
    for (int f = 0; f < 4; ++f)
#pragma unroll
      for (int rr = 0; rr < 4; ++rr) {
        ex[f][rr] = __expf(ctx[f][g * 4 + rr] - mx);
        se += ex[f][rr];
      }
    se += __shfl_xor(se, 1);
    se += __shfl_xor(se, 2);
    se += __shfl_xor(se, 4);
    se += __shfl_xor(se, 8);
    float inv = 1.0f / se;
#pragma unroll
    for (int f = 0; f < 4; ++f)
#pragma unroll
      for (int rr = 0; rr < 4; ++rr) {
        int mrow = g * 4 + rr + hi * 8;
        int path = pbase + mrow;
        out[(path >> 2) * 256 + (path & 3) * 64 + f * 16 + lane15] =
            ex[f][rr] * inv;
      }
  }
}

}  // namespace

extern "C" void kernel_launch(void* const* d_in, const int* in_sizes, int n_in,
                              void* d_out, int out_size, void* d_ws, size_t ws_size,
                              hipStream_t stream) {
  (void)in_sizes; (void)n_in; (void)out_size; (void)d_ws; (void)ws_size;
  const float* link_states = (const float*)d_in[0];
  const float* Wx          = (const float*)d_in[1];
  const float* Wh          = (const float*)d_in[2];
  const float* b           = (const float*)d_in[3];
  const float* wq          = (const float*)d_in[4];
  const float* wk          = (const float*)d_in[5];
  const float* wv          = (const float*)d_in[6];
  const int*   paths       = (const int*)d_in[7];
  // d_in[8]=idx, d_in[9]=seqs are structurally determined (b*16+t) -> unused.
  float* out = (float*)d_out;

  dim3 grid(B_PATHS / (WAVES * 16));  // 256 blocks x 128 threads (4 waves)
  path_embed_kernel<<<grid, TPB, 0, stream>>>(link_states, Wx, Wh, b, wq, wk, wv,
                                              paths, out);
}
